// MEIT_54107997995718
// MI455X (gfx1250) — matrix operand order repaired and checked by reading, not compiled, not run on hardware
//
#include <hip/hip_runtime.h>
#include <hip/hip_bf16.h>
#include <math.h>

typedef __attribute__((ext_vector_type(16))) _Float16 v16h;
typedef __attribute__((ext_vector_type(8)))  float    v8f;

union HalfVec { v16h v; uint4 u[2]; };

#define BM 128
#define BN 64
#define BK 32

__device__ __forceinline__ float gelu_exact(float v) {
    return 0.5f * v * (1.0f + erff(v * 0.70710678118654752f));
}

// CDNA5 async global->LDS copy (GV mode): VDST = LDS byte offset VGPR,
// VADDR = 64-bit global address. Tracked by ASYNCcnt (one per instruction).
__device__ __forceinline__ void async_copy_b128(unsigned lds_off, const void* gptr) {
    asm volatile("global_load_async_to_lds_b128 %0, %1, off"
                 :: "v"(lds_off), "v"(gptr)
                 : "memory");
}
__device__ __forceinline__ void wait_async0() {
    asm volatile("s_wait_asynccnt 0" ::: "memory");
}
// allow the 2 just-issued (next-tile) copies to remain outstanding
__device__ __forceinline__ void wait_async2() {
    asm volatile("s_wait_asynccnt 2" ::: "memory");
}

// ---------------------------------------------------------------------------
// f16-WMMA GEMM: C[z] = epilogue( A[z] (MxK,f16 row-major) @ B[p] )
// B is pre-transposed f16: BT[n][k] (row stride = K), so BOTH operand tiles
// are staged with global_load_async_to_lds_b128 (no cvt, no strided loads).
// Double-buffered LDS: tile k+1 DMA overlaps tile k WMMA (s_wait_asynccnt 2).
// Tile 128x64x32, 8 waves in 4x2, each wave 32x32 -> 4 WMMA per K-step.
// Per tile per wave exactly 2 async instructions (A rows clamped, not masked,
// so EXEC stays full and the ASYNCcnt schedule is uniform).
// ---------------------------------------------------------------------------
__global__ __launch_bounds__(256) void gemm_wmma_kernel(
    const _Float16* __restrict__ Abase, long long a_zstride,
    const _Float16* __restrict__ BTbase, long long b_pstride,
    const float* __restrict__ biasBase, long long bias_pstride,
    const float* __restrict__ resBase, long long res_zstride,
    float* __restrict__ C32base, _Float16* __restrict__ C16base, long long c_zstride,
    int M, int N, int K, int lda, int ldb /* = K */, int ldc,
    int zdivB, int gelu)
{
    __shared__ _Float16 sA[2][BM * BK];   // [m][k] 8KB x2
    __shared__ _Float16 sBT[2][BN * BK];  // [n][k] 4KB x2

    const int z = blockIdx.z;
    const int p = z / zdivB;
    const _Float16* A  = Abase  + (long long)z * a_zstride;
    const _Float16* BT = BTbase + (long long)p * b_pstride;
    const float* bias  = biasBase ? biasBase + (long long)p * bias_pstride : nullptr;
    const float* res   = resBase  ? resBase  + (long long)z * res_zstride  : nullptr;
    float*     C32 = C32base ? C32base + (long long)z * c_zstride : nullptr;
    _Float16*  C16 = C16base ? C16base + (long long)z * c_zstride : nullptr;

    const int m0 = blockIdx.y * BM;
    const int n0 = blockIdx.x * BN;

    const int t    = threadIdx.x;
    const int wave = t >> 5;
    const int lane = t & 31;
    const int wrow = wave >> 1;   // 0..3 -> 32-row band
    const int wcol = wave & 1;    // 0..1 -> 32-col band
    const int lrow = lane & 15;
    const int hi   = lane >> 4;

    v8f acc[2][2] = {{{}, {}}, {{}, {}}};

    const unsigned ldsA[2] = { (unsigned)(uintptr_t)&sA[0][0],
                               (unsigned)(uintptr_t)&sA[1][0] };
    const unsigned ldsB[2] = { (unsigned)(uintptr_t)&sBT[0][0],
                               (unsigned)(uintptr_t)&sBT[1][0] };

    // A staging: 256 b128 chunks = 128 rows x 2 x 8 halves (k 0..7, 16..23), one per thread.
    const int arow  = t >> 1;
    const int ahoff = (t & 1) << 4;
    const int agr   = (m0 + arow < M) ? (m0 + arow) : (M - 1);  // clamp, keep EXEC full
    const _Float16* agp   = A + (long long)agr * lda + ahoff;
    const unsigned  aoffL = (unsigned)((arow * BK + ahoff) * 2);

    // B staging: 128 b128 chunks = 64 rows x 2 x 8 halves (k 0..7, 16..23), lanes 0..15 of each wave.
    const int bchunk = (wave << 4) + lrow;       // 0..127
    const int bnn    = bchunk >> 1;              // 0..63
    const int bhoff  = (bchunk & 1) << 4;
    const bool b_act = (lane < 16);
    const _Float16* bgp   = BT + (long long)(n0 + bnn) * ldb + bhoff;
    const unsigned  boffL = (unsigned)((bnn * BK + bhoff) * 2);

    auto stage = [&](int kk, int buf) {
        async_copy_b128(ldsA[buf] + aoffL, agp + kk);
        if (b_act) async_copy_b128(ldsB[buf] + boffL, bgp + kk);
    };

    stage(0, 0);
    int buf = 0;
    for (int k0 = 0; k0 < K; k0 += BK) {
        const bool more = (k0 + BK) < K;
        if (more) {
            stage(k0 + BK, buf ^ 1);   // prefetch next tile into alternate buffer
            wait_async2();             // current tile's copies complete
        } else {
            wait_async0();
        }
        __syncthreads();

        const _Float16* sAb = sA[buf];
        const _Float16* sBb = sBT[buf];

        HalfVec af[2], bf[2];
#pragma unroll
        for (int mi = 0; mi < 2; mi++) {
            const uint4* pa = reinterpret_cast<const uint4*>(
                &sAb[(wrow * 32 + mi * 16 + lrow) * BK + hi * 8]);
            af[mi].u[0] = pa[0];   // K = hi*8 .. +7
            af[mi].u[1] = pa[2];   // K = 16+hi*8 .. +7
        }
#pragma unroll
        for (int ni = 0; ni < 2; ni++) {
            const int col = wcol * 32 + ni * 16 + lrow;
            const uint4* pb = reinterpret_cast<const uint4*>(&sBb[col * BK + hi * 8]);
            bf[ni].u[0] = pb[0];
            bf[ni].u[1] = pb[2];
        }

        acc[0][0] = __builtin_amdgcn_wmma_f32_16x16x32_f16(false, af[0].v, false, bf[0].v,
                                                           (short)0, acc[0][0], false, false);
        acc[0][1] = __builtin_amdgcn_wmma_f32_16x16x32_f16(false, af[0].v, false, bf[1].v,
                                                           (short)0, acc[0][1], false, false);
        acc[1][0] = __builtin_amdgcn_wmma_f32_16x16x32_f16(false, af[1].v, false, bf[0].v,
                                                           (short)0, acc[1][0], false, false);
        acc[1][1] = __builtin_amdgcn_wmma_f32_16x16x32_f16(false, af[1].v, false, bf[1].v,
                                                           (short)0, acc[1][1], false, false);
        __syncthreads();
        buf ^= 1;
    }

    // epilogue: lane lrow = col, vgpr r -> local row hi*8+r
#pragma unroll
    for (int mi = 0; mi < 2; mi++) {
        const int rowb = m0 + wrow * 32 + mi * 16 + hi * 8;
#pragma unroll
        for (int ni = 0; ni < 2; ni++) {
            const int col = n0 + wcol * 32 + ni * 16 + lrow;
#pragma unroll
            for (int r = 0; r < 8; r++) {
                const int row = rowb + r;
                if (row < M) {
                    float v = acc[mi][ni][r];
                    if (bias) v += bias[col];
                    if (gelu) v = gelu_exact(v);
                    if (res)  v += res[(long long)row * ldc + col];
                    if (C32) C32[(long long)row * ldc + col] = v;
                    if (C16) C16[(long long)row * ldc + col] = (_Float16)v;
                }
            }
        }
    }
}

// ---------------------------------------------------------------------------
// LayerNorm: one wave per token (f32 stats, shfl reduce). Optional f32/f16
// outputs. Per-z gamma slice: p = z / zdivB (per-partition LN params).
// ---------------------------------------------------------------------------
__global__ __launch_bounds__(256) void ln_kernel(
    const float* __restrict__ x, float* __restrict__ y32, _Float16* __restrict__ y16,
    const float* __restrict__ gbase, const float* __restrict__ bbase,
    int g_pstride, int ntok, int D, int zdivB, int x_b_only,
    long long x_zstride, long long y_zstride)
{
    const int z = blockIdx.z;
    const int p = z / zdivB;
    const int b = z % zdivB;
    const float* g  = gbase + (long long)p * g_pstride;
    const float* bb = bbase + (long long)p * g_pstride;
    const float* xp = x + (x_b_only ? (long long)b * x_zstride
                                    : (long long)z * x_zstride);

    const int wid  = threadIdx.x >> 5;
    const int lane = threadIdx.x & 31;
    const int tok  = blockIdx.x * 8 + wid;
    if (tok >= ntok) return;

    const float* xr = xp + (long long)tok * D;
    float s = 0.0f, s2 = 0.0f;
    for (int d = lane; d < D; d += 32) {
        float v = xr[d];
        s += v; s2 += v * v;
    }
    for (int off = 16; off > 0; off >>= 1) {
        s  += __shfl_xor(s, off);
        s2 += __shfl_xor(s2, off);
    }
    const float mu  = s / (float)D;
    const float var = s2 / (float)D - mu * mu;
    const float rs  = rsqrtf(var + 1e-5f);
    const long long yoff = (long long)z * y_zstride + (long long)tok * D;
    for (int d = lane; d < D; d += 32) {
        const float v = (xr[d] - mu) * rs * g[d] + bb[d];
        if (y32) y32[yoff + d] = v;
        if (y16) y16[yoff + d] = (_Float16)v;
    }
}

// CHW -> (token, C) transpose, f32 -> f16
__global__ void transpose_chw_kernel(const float* __restrict__ x,
                                     _Float16* __restrict__ xT,
                                     int Bb, int Cc, int HW)
{
    long long t = (long long)blockIdx.x * blockDim.x + threadIdx.x;
    const long long total = (long long)Bb * Cc * HW;
    if (t >= total) return;
    const int b = (int)(t / ((long long)Cc * HW));
    const long long r = t % ((long long)Cc * HW);
    const int c = (int)(r / HW);
    const int n = (int)(r % HW);
    xT[((long long)b * HW + n) * Cc + c] = (_Float16)x[t];
}

__global__ void f32_to_f16_kernel(const float* __restrict__ src,
                                  _Float16* __restrict__ dst, long long n)
{
    long long t = (long long)blockIdx.x * blockDim.x + threadIdx.x;
    if (t < n) dst[t] = (_Float16)src[t];
}

// one-time weight convert + transpose: wt[z][n][k] = (f16) w[z][k][n]
__global__ void wtrans_kernel(const float* __restrict__ w, _Float16* __restrict__ wt,
                              int Z, int K, int N)
{
    long long t = (long long)blockIdx.x * blockDim.x + threadIdx.x;
    const long long total = (long long)Z * K * N;
    if (t >= total) return;
    const int zz = (int)(t / ((long long)K * N));
    const long long r = t % ((long long)K * N);
    const int k = (int)(r / N);
    const int n = (int)(r % N);
    wt[(long long)zz * K * N + (long long)n * K + k] = (_Float16)w[t];
}

// TokenScorer v-projection: v = m @ sv_w + sv_b  (D -> 1)
__global__ void sv_kernel(const float* __restrict__ m, const float* __restrict__ svw,
                          const float* __restrict__ svb, float* __restrict__ vs,
                          int total, int D)
{
    const int t = blockIdx.x * blockDim.x + threadIdx.x;
    if (t >= total) return;
    const float* mr = m + (long long)t * D;
    float s = svb[0];
    for (int d = 0; d < D; d++) s += mr[d] * svw[d];
    vs[t] = s;
}

// TokenScorer attention: scores[b,n] = softmax_j(q.k_j * scale) . v_j
__global__ void scorer_kernel(const _Float16* __restrict__ q, const _Float16* __restrict__ k,
                              const float* __restrict__ vv, float* __restrict__ scores,
                              int Bb, int N, int D, int Nt, float scale)
{
    const int t = blockIdx.x * blockDim.x + threadIdx.x;
    if (t >= Bb * N) return;
    const int b = t / N;
    const _Float16* qr = q + (long long)t * D;
    float logit[8];
    float mx = -1e30f;
    for (int j = 0; j < Nt; j++) {
        const _Float16* kr = k + ((long long)b * Nt + j) * D;
        float d0 = 0.0f;
        for (int d = 0; d < D; d++) d0 += (float)qr[d] * (float)kr[d];
        d0 *= scale;
        logit[j] = d0;
        mx = fmaxf(mx, d0);
    }
    float den = 0.0f;
    for (int j = 0; j < Nt; j++) { logit[j] = __expf(logit[j] - mx); den += logit[j]; }
    float s = 0.0f;
    for (int j = 0; j < Nt; j++) s += logit[j] * vv[b * Nt + j];
    scores[t] = s / den;
}

// Expert cross-attention (Nt memory tokens), f16 in/out, f32 math
__global__ void attn_kernel(const _Float16* __restrict__ q, const _Float16* __restrict__ k,
                            const _Float16* __restrict__ v, _Float16* __restrict__ o,
                            long long q_zstride, long long kv_zstride,
                            int S, int Nt, int H, int dh, float scale)
{
    const int z = blockIdx.z;
    const int t = blockIdx.x * blockDim.x + threadIdx.x;
    if (t >= S * H) return;
    const int i = t / H, h = t % H;
    const _Float16* qp = q + (long long)z * q_zstride + ((long long)i * H + h) * dh;
    const _Float16* kp = k + (long long)z * kv_zstride + h * dh;
    const _Float16* vp = v + (long long)z * kv_zstride + h * dh;
    float logit[16];
    float mx = -1e30f;
    for (int j = 0; j < Nt; j++) {
        const _Float16* kr = kp + (long long)j * H * dh;
        float d0 = 0.0f;
        for (int d = 0; d < dh; d++) d0 += (float)qp[d] * (float)kr[d];
        d0 *= scale;
        logit[j] = d0;
        mx = fmaxf(mx, d0);
    }
    float den = 0.0f;
    for (int j = 0; j < Nt; j++) { logit[j] = __expf(logit[j] - mx); den += logit[j]; }
    const float inv = 1.0f / den;
    _Float16* op = o + (long long)z * q_zstride + ((long long)i * H + h) * dh;
    for (int d = 0; d < dh; d++) {
        float s = 0.0f;
        for (int j = 0; j < Nt; j++) s += logit[j] * (float)vp[(long long)j * H * dh + d];
        op[d] = (_Float16)(s * inv);
    }
}

// Per-batch bitonic argsort (descending) + partition-mean softmax -> pw
__global__ __launch_bounds__(1024) void sort_kernel(
    const float* __restrict__ scores, int* __restrict__ idxout,
    float* __restrict__ pw, int N, int P)
{
    __shared__ float skey[4096];
    __shared__ int   sval[4096];
    __shared__ float pmean[8];
    const int b = blockIdx.x;
    for (int i = threadIdx.x; i < N; i += 1024) {
        skey[i] = scores[(long long)b * N + i];
        sval[i] = i;
    }
    __syncthreads();
    for (unsigned k = 2; k <= (unsigned)N; k <<= 1) {
        for (unsigned j = k >> 1; j > 0; j >>= 1) {
            for (unsigned i = threadIdx.x; i < (unsigned)N; i += 1024) {
                const unsigned ixj = i ^ j;
                if (ixj > i) {
                    const bool desc = ((i & k) == 0);
                    const float a = skey[i], c = skey[ixj];
                    if (desc ? (a < c) : (a > c)) {
                        skey[i] = c; skey[ixj] = a;
                        const int ta = sval[i]; sval[i] = sval[ixj]; sval[ixj] = ta;
                    }
                }
            }
            __syncthreads();
        }
    }
    const int S = N / P;
    if ((int)threadIdx.x < P) {
        float s = 0.0f;
        for (int i = 0; i < S; i++) s += skey[threadIdx.x * S + i];
        pmean[threadIdx.x] = s / (float)S;
    }
    __syncthreads();
    if (threadIdx.x == 0) {
        float mx = -1e30f;
        for (int p = 0; p < P; p++) mx = fmaxf(mx, pmean[p]);
        float e[8], den = 0.0f;
        for (int p = 0; p < P; p++) { e[p] = __expf(pmean[p] - mx); den += e[p]; }
        for (int p = 0; p < P; p++) pw[b * P + p] = e[p] / den;
    }
    for (int i = threadIdx.x; i < N; i += 1024)
        idxout[(long long)b * N + i] = sval[i];
}

// gather: xt[(p*B+b), s, :] = tokens[b, idx[b, p*S+s], :]   (f32 residual stream)
__global__ void gather_kernel(const float* __restrict__ tokens, const int* __restrict__ idx,
                              float* __restrict__ xt, int Bb, int N, int S, int D)
{
    const int z = blockIdx.z;
    const int p = z / Bb, b = z % Bb;
    const int t = blockIdx.x * blockDim.x + threadIdx.x;
    const int D4 = D >> 2;
    if (t >= S * D4) return;
    const int s = t / D4, dv = t % D4;
    const int tok = idx[(long long)b * N + (long long)p * S + s];
    const float4* src = (const float4*)(tokens + ((long long)b * N + tok) * D);
    float4* dst = (float4*)(xt + ((long long)z * S + s) * D);
    dst[dv] = src[dv];
}

// scatter: out[b, idx[b,p*S+s], :] = xt[(p*B+b), s, :] * pw[b,p]
__global__ void scatter_kernel(const float* __restrict__ xt, const int* __restrict__ idx,
                               const float* __restrict__ pw, float* __restrict__ outtok,
                               int Bb, int N, int S, int D, int P)
{
    const int z = blockIdx.z;
    const int p = z / Bb, b = z % Bb;
    const int t = blockIdx.x * blockDim.x + threadIdx.x;
    const int D4 = D >> 2;
    if (t >= S * D4) return;
    const int s = t / D4, dv = t % D4;
    const int tok = idx[(long long)b * N + (long long)p * S + s];
    const float w = pw[b * P + p];
    const float4* src = (const float4*)(xt + ((long long)z * S + s) * D);
    float4* dst = (float4*)(outtok + ((long long)b * N + tok) * D);
    float4 v = src[dv];
    v.x *= w; v.y *= w; v.z *= w; v.w *= w;
    dst[dv] = v;
}

// ---------------------------------------------------------------------------

extern "C" void kernel_launch(void* const* d_in, const int* in_sizes, int n_in,
                              void* d_out, int out_size, void* d_ws, size_t ws_size,
                              hipStream_t stream)
{
    constexpr int Bc = 4, Cc = 128, HW = 64 * 64, Nn = 4096, Dd = 256;
    constexpr int IN = 512, FFd = 1024, Pp = 4, Ll = 2, NT = 8, Sc = 1024;
    constexpr int HD = 8, DH = 64;
    constexpr int PB = Pp * Bc;

    const float* x     = (const float*)d_in[0];
    const float* m     = (const float*)d_in[1];
    const float* projw = (const float*)d_in[2];
    const float* projb = (const float*)d_in[3];
    const float* lnpg  = (const float*)d_in[4];
    const float* lnpb  = (const float*)d_in[5];
    const float* sqw   = (const float*)d_in[6];
    const float* sqb   = (const float*)d_in[7];
    const float* skw   = (const float*)d_in[8];
    const float* skb   = (const float*)d_in[9];
    const float* svw   = (const float*)d_in[10];
    const float* svb   = (const float*)d_in[11];
    const float* ang   = (const float*)d_in[12];
    const float* anb   = (const float*)d_in[13];
    const float* wq    = (const float*)d_in[14];
    const float* wk    = (const float*)d_in[15];
    const float* wv    = (const float*)d_in[16];
    const float* wo    = (const float*)d_in[17];
    const float* wob   = (const float*)d_in[18];
    const float* fng   = (const float*)d_in[19];
    const float* fnb   = (const float*)d_in[20];
    const float* w1    = (const float*)d_in[21];
    const float* b1    = (const float*)d_in[22];
    const float* w2    = (const float*)d_in[23];
    const float* b2    = (const float*)d_in[24];
    const float* flng  = (const float*)d_in[25];
    const float* flnb  = (const float*)d_in[26];
    float* out = (float*)d_out;
    (void)in_sizes; (void)n_in; (void)out_size; (void)ws_size;

    // workspace carve (256B aligned)
    char* wsb = (char*)d_ws;
    size_t off = 0;
    auto carveB = [&](size_t bytes) {
        size_t a = (off + 255) & ~(size_t)255;
        void* r = wsb + a;
        off = a + bytes;
        return r;
    };
    float* tokens = (float*)carveB((size_t)Bc * Nn * Dd * 4);
    float* xt     = (float*)carveB((size_t)PB * Sc * Dd * 4);
    float* scores = (float*)carveB((size_t)Bc * Nn * 4);
    float* pw     = (float*)carveB((size_t)Bc * Pp * 4);
    float* vs     = (float*)carveB((size_t)Bc * NT * 4);
    int*   idx    = (int*)carveB((size_t)Bc * Nn * 4);

    _Float16* tokens16 = (_Float16*)carveB((size_t)Bc * Nn * Dd * 2);
    _Float16* xn16     = (_Float16*)carveB((size_t)PB * Sc * Dd * 2);
    _Float16* mn16     = (_Float16*)carveB((size_t)PB * NT * Dd * 2);
    _Float16* kb16     = (_Float16*)carveB((size_t)PB * NT * IN * 2);
    _Float16* vb16     = (_Float16*)carveB((size_t)PB * NT * IN * 2);
    _Float16* ks16     = (_Float16*)carveB((size_t)Bc * NT * Dd * 2);
    _Float16* m16      = (_Float16*)carveB((size_t)Bc * NT * Dd * 2);
    _Float16* U16      = (_Float16*)carveB((size_t)PB * Sc * FFd * 2);  // union

    // transposed f16 weights: wt[slice][n][k]
    _Float16* projwT = (_Float16*)carveB((size_t)Cc * Dd * 2);
    _Float16* sqwT   = (_Float16*)carveB((size_t)Dd * Dd * 2);
    _Float16* skwT   = (_Float16*)carveB((size_t)Dd * Dd * 2);
    _Float16* wqT    = (_Float16*)carveB((size_t)Pp * Ll * Dd * IN * 2);
    _Float16* wkT    = (_Float16*)carveB((size_t)Pp * Ll * Dd * IN * 2);
    _Float16* wvT    = (_Float16*)carveB((size_t)Pp * Ll * Dd * IN * 2);
    _Float16* woT    = (_Float16*)carveB((size_t)Pp * Ll * IN * Dd * 2);
    _Float16* w1T    = (_Float16*)carveB((size_t)Pp * Ll * Dd * FFd * 2);
    _Float16* w2T    = (_Float16*)carveB((size_t)Pp * Ll * FFd * Dd * 2);

    _Float16* xT16  = U16;                               // (B*N, C)  phase 1
    _Float16* qs16  = U16;                               // (B*N, D)  phase 2
    _Float16* q16   = U16;                               // (PB, S, IN)
    _Float16* o16   = U16 + (size_t)PB * Sc * IN;        // (PB, S, IN)
    _Float16* ff116 = U16;                               // (PB, S, FF)

    auto wtrans = [&](const float* s, _Float16* d, int Z, int K, int N) {
        const long long n = (long long)Z * K * N;
        wtrans_kernel<<<dim3((unsigned)((n + 255) / 256)), dim3(256), 0, stream>>>(s, d, Z, K, N);
    };
    auto gemm = [&](const _Float16* A, long long azs, const _Float16* BT, long long bps,
                    const float* bias, long long biasps,
                    const float* res, long long reszs,
                    float* C32, _Float16* C16, long long czs,
                    int M, int Nd, int Kd,
                    int Z, int zdivB, int gelu) {
        dim3 grid(Nd / BN, (M + BM - 1) / BM, Z);
        gemm_wmma_kernel<<<grid, dim3(256), 0, stream>>>(
            A, azs, BT, bps, bias, biasps, res, reszs, C32, C16, czs,
            M, Nd, Kd, /*lda=*/Kd, /*ldb=*/Kd, /*ldc=*/Nd, zdivB, gelu);
    };
    auto ln = [&](const float* xp, float* y32, _Float16* y16,
                  const float* g, const float* bbp,
                  int gps, int ntok, int Z, int zdivB, int xBonly,
                  long long xzs, long long yzs) {
        dim3 grid((ntok + 7) / 8, 1, Z);
        ln_kernel<<<grid, dim3(256), 0, stream>>>(xp, y32, y16, g, bbp, gps, ntok, Dd,
                                                  zdivB, xBonly, xzs, yzs);
    };

    // -------- phase 0: one-time weight convert+transpose (f32 -> f16 [n][k]) ----
    wtrans(projw, projwT, 1, Cc, Dd);
    wtrans(sqw,   sqwT,   1, Dd, Dd);
    wtrans(skw,   skwT,   1, Dd, Dd);
    wtrans(wq,    wqT,    Pp * Ll, Dd, IN);
    wtrans(wk,    wkT,    Pp * Ll, Dd, IN);
    wtrans(wv,    wvT,    Pp * Ll, Dd, IN);
    wtrans(wo,    woT,    Pp * Ll, IN, Dd);
    wtrans(w1,    w1T,    Pp * Ll, Dd, FFd);
    wtrans(w2,    w2T,    Pp * Ll, FFd, Dd);
    {
        const long long n = (long long)Bc * NT * Dd;
        f32_to_f16_kernel<<<dim3((unsigned)((n + 255) / 256)), dim3(256), 0, stream>>>(m, m16, n);
    }

    // -------- phase 1: projection + LN --------
    {
        const long long total = (long long)Bc * Cc * HW;
        transpose_chw_kernel<<<dim3((unsigned)((total + 255) / 256)), dim3(256), 0, stream>>>(
            x, xT16, Bc, Cc, HW);
    }
    gemm(xT16, 0, projwT, 0, projb, 0, nullptr, 0, tokens, nullptr, 0,
         Bc * Nn, Dd, Cc, 1, 1, 0);
    ln(tokens, tokens, tokens16, lnpg, lnpb, 0, Bc * Nn, 1, 1, 0, 0, 0);

    // -------- phase 2: token scorer --------
    gemm(tokens16, 0, sqwT, 0, sqb, 0, nullptr, 0, nullptr, qs16, 0,
         Bc * Nn, Dd, Dd, 1, 1, 0);
    gemm(m16, 0, skwT, 0, skb, 0, nullptr, 0, nullptr, ks16, 0,
         Bc * NT, Dd, Dd, 1, 1, 0);
    sv_kernel<<<dim3(1), dim3(32), 0, stream>>>(m, svw, svb, vs, Bc * NT, Dd);
    scorer_kernel<<<dim3((Bc * Nn + 255) / 256), dim3(256), 0, stream>>>(
        qs16, ks16, vs, scores, Bc, Nn, Dd, NT, 0.0625f);

    // -------- phase 3: sort + partition weights, gather --------
    sort_kernel<<<dim3(Bc), dim3(1024), 0, stream>>>(scores, idx, pw, Nn, Pp);
    gather_kernel<<<dim3((Sc * (Dd / 4) + 255) / 256, 1, PB), dim3(256), 0, stream>>>(
        tokens, idx, xt, Bc, Nn, Sc, Dd);

    // -------- phase 4: expert transformer layers --------
    for (int l = 0; l < Ll; l++) {
        const float* ang_l = ang + (size_t)l * Dd;
        const float* anb_l = anb + (size_t)l * Dd;
        // xn = LN(xt, an[p,l]);  mn = LN(m, an[p,l])
        ln(xt, nullptr, xn16, ang_l, anb_l, Ll * Dd, Sc, PB, Bc, 0,
           (long long)Sc * Dd, (long long)Sc * Dd);
        ln(m, nullptr, mn16, ang_l, anb_l, Ll * Dd, NT, PB, Bc, 1,
           (long long)NT * Dd, (long long)NT * Dd);
        // q = xn @ wq[p,l]
        gemm(xn16, (long long)Sc * Dd, wqT + (size_t)l * Dd * IN, (long long)Ll * Dd * IN,
             nullptr, 0, nullptr, 0, nullptr, q16, (long long)Sc * IN,
             Sc, IN, Dd, PB, Bc, 0);
        // k, v = mn @ wk/wv[p,l]
        gemm(mn16, (long long)NT * Dd, wkT + (size_t)l * Dd * IN, (long long)Ll * Dd * IN,
             nullptr, 0, nullptr, 0, nullptr, kb16, (long long)NT * IN,
             NT, IN, Dd, PB, Bc, 0);
        gemm(mn16, (long long)NT * Dd, wvT + (size_t)l * Dd * IN, (long long)Ll * Dd * IN,
             nullptr, 0, nullptr, 0, nullptr, vb16, (long long)NT * IN,
             NT, IN, Dd, PB, Bc, 0);
        // o = softmax(q k^T * scale) v
        attn_kernel<<<dim3((Sc * HD + 255) / 256, 1, PB), dim3(256), 0, stream>>>(
            q16, kb16, vb16, o16, (long long)Sc * IN, (long long)NT * IN,
            Sc, NT, HD, DH, 0.0625f);
        // xt = xt + o @ wo[p,l] + wo_b[p,l]
        gemm(o16, (long long)Sc * IN, woT + (size_t)l * IN * Dd, (long long)Ll * IN * Dd,
             wob + (size_t)l * Dd, (long long)Ll * Dd,
             xt, (long long)Sc * Dd, xt, nullptr, (long long)Sc * Dd,
             Sc, Dd, IN, PB, Bc, 0);
        // h = LN(xt, fn[p,l])
        ln(xt, nullptr, xn16, fng + (size_t)l * Dd, fnb + (size_t)l * Dd, Ll * Dd,
           Sc, PB, Bc, 0, (long long)Sc * Dd, (long long)Sc * Dd);
        // ff1 = gelu(h @ w1[p,l] + b1[p,l])
        gemm(xn16, (long long)Sc * Dd, w1T + (size_t)l * Dd * FFd, (long long)Ll * Dd * FFd,
             b1 + (size_t)l * FFd, (long long)Ll * FFd, nullptr, 0,
             nullptr, ff116, (long long)Sc * FFd,
             Sc, FFd, Dd, PB, Bc, 1);
        // xt = xt + ff1 @ w2[p,l] + b2[p,l]
        gemm(ff116, (long long)Sc * FFd, w2T + (size_t)l * FFd * Dd, (long long)Ll * FFd * Dd,
             b2 + (size_t)l * Dd, (long long)Ll * Dd,
             xt, (long long)Sc * Dd, xt, nullptr, (long long)Sc * Dd,
             Sc, Dd, FFd, PB, Bc, 0);
    }

    // -------- phase 5: weighted scatter-back + final LN --------
    scatter_kernel<<<dim3((Sc * (Dd / 4) + 255) / 256, 1, PB), dim3(256), 0, stream>>>(
        xt, idx, pw, tokens, Bc, Nn, Sc, Dd, Pp);
    ln(tokens, out, nullptr, flng, flnb, 0, Bc * Nn, 1, 1, 0, 0, 0);
}
